// HAN_7627861918395
// MI455X (gfx1250) — matrix operand order repaired and checked by reading, not compiled, not run on hardware
//
#include <hip/hip_runtime.h>
#include <hip/hip_bf16.h>

#define H 2
#define D 128
#define F_OUT 256
#define N_NODE 20000
#define E_EDGE 200000
#define N_SEL 4096
#define K_IN 768

typedef __attribute__((ext_vector_type(16))) __bf16 v16bf;
typedef __attribute__((ext_vector_type(8)))  float  v8f;
typedef unsigned short ushort_t;

struct uint4p { unsigned int x, y, z, w; };           // POD 16B for b128 loads
union Frag16 { v16bf v; uint4p q[2]; ushort_t s[16]; };
union Pack8  { ushort_t s[8]; uint4p q; };

__device__ __forceinline__ ushort_t f2bf(float f) {
    unsigned int u = __float_as_uint(f);
    unsigned int r = u + 0x7FFFu + ((u >> 16) & 1u);   // round-to-nearest-even
    return (ushort_t)(r >> 16);
}

__device__ __forceinline__ void pack8(ushort_t* s, float4 a, float4 b) {
    s[0] = f2bf(a.x); s[1] = f2bf(a.y); s[2] = f2bf(a.z); s[3] = f2bf(a.w);
    s[4] = f2bf(b.x); s[5] = f2bf(b.y); s[6] = f2bf(b.z); s[7] = f2bf(b.w);
}

__device__ __forceinline__ void atomicMaxFloat(float* addr, float val) {
    if (val >= 0.0f) atomicMax((int*)addr, __float_as_int(val));
    else             atomicMin((unsigned int*)addr, __float_as_uint(val));
}

// ---------------------------------------------------------------------------
__global__ void fill_u32_kernel(unsigned int* __restrict__ p, unsigned int val, size_t n) {
    size_t i = (size_t)blockIdx.x * blockDim.x + threadIdx.x;
    size_t stride = (size_t)gridDim.x * blockDim.x;
    for (; i < n; i += stride) p[i] = val;
}

// f32 -> bf16 (RNE), 8 elements / thread, vectorized b128 store
__global__ void cvt_bf16_kernel(const float* __restrict__ src,
                                ushort_t* __restrict__ dst, size_t n) {
    size_t i = ((size_t)blockIdx.x * blockDim.x + threadIdx.x) * 8;
    size_t stride = (size_t)gridDim.x * blockDim.x * 8;
    for (; i < n; i += stride) {
        float4 a = *(const float4*)(src + i);
        float4 b = *(const float4*)(src + i + 4);
        Pack8 p;
        pack8(p.s, a, b);
        *(uint4p*)(dst + i) = p.q;
    }
}

// relu in place (f32) + bf16 copy for the semantic-attention GEMM
__global__ void relu_cvt_kernel(float* __restrict__ x,
                                ushort_t* __restrict__ xb, size_t n) {
    size_t i = ((size_t)blockIdx.x * blockDim.x + threadIdx.x) * 8;
    size_t stride = (size_t)gridDim.x * blockDim.x * 8;
    for (; i < n; i += stride) {
        float4 a = *(const float4*)(x + i);
        float4 b = *(const float4*)(x + i + 4);
        a.x = fmaxf(a.x, 0.f); a.y = fmaxf(a.y, 0.f); a.z = fmaxf(a.z, 0.f); a.w = fmaxf(a.w, 0.f);
        b.x = fmaxf(b.x, 0.f); b.y = fmaxf(b.y, 0.f); b.z = fmaxf(b.z, 0.f); b.w = fmaxf(b.w, 0.f);
        *(float4*)(x + i)     = a;
        *(float4*)(x + i + 4) = b;
        Pack8 p;
        pack8(p.s, a, b);
        *(uint4p*)(xb + i) = p.q;
    }
}

// ---------------------------------------------------------------------------
// Y[M,256] = Xb[M,K] @ Wb[256,K]^T + bias ; bf16 inputs, f32 accum
// one wave computes a 16x64 tile: 1 A-frag, 4 B-frags, 4 WMMAs per K-step
__global__ void __launch_bounds__(256) gemm_bf_kernel(
    const ushort_t* __restrict__ Xb, const ushort_t* __restrict__ Wb,
    const float* __restrict__ bias, float* __restrict__ Y, int M, int K)
{
    int wave = (int)((blockIdx.x * blockDim.x + threadIdx.x) >> 5);
    int lane = threadIdx.x & 31;
    int tm  = wave >> 2;           // F_OUT/64 = 4 N-blocks
    int tn4 = wave & 3;
    if (tm * 16 >= M) return;
    int half = lane >> 4;
    int l15  = lane & 15;
    const ushort_t* arow = Xb + (size_t)(tm * 16 + l15) * K;
    const ushort_t* br0  = Wb + (size_t)(tn4 * 64 + 0  + l15) * K;
    const ushort_t* br1  = Wb + (size_t)(tn4 * 64 + 16 + l15) * K;
    const ushort_t* br2  = Wb + (size_t)(tn4 * 64 + 32 + l15) * K;
    const ushort_t* br3  = Wb + (size_t)(tn4 * 64 + 48 + l15) * K;
    v8f acc0 = {}, acc1 = {}, acc2 = {}, acc3 = {};
    for (int k0 = 0; k0 < K; k0 += 32) {
        Frag16 a, b0, b1, b2, b3;
        // A: row l15, K = k0 + half*16 + {0..15}
        a.q[0] = *(const uint4p*)(arow + k0 + half * 16);
        a.q[1] = *(const uint4p*)(arow + k0 + half * 16 + 8);
        // B: col l15 (of each 16-wide subtile), K = k0 + half*16 + {0..15}
        int kb = k0 + half * 16;
        b0.q[0] = *(const uint4p*)(br0 + kb); b0.q[1] = *(const uint4p*)(br0 + kb + 8);
        b1.q[0] = *(const uint4p*)(br1 + kb); b1.q[1] = *(const uint4p*)(br1 + kb + 8);
        b2.q[0] = *(const uint4p*)(br2 + kb); b2.q[1] = *(const uint4p*)(br2 + kb + 8);
        b3.q[0] = *(const uint4p*)(br3 + kb); b3.q[1] = *(const uint4p*)(br3 + kb + 8);
        acc0 = __builtin_amdgcn_wmma_f32_16x16x32_bf16(false, a.v, false, b0.v, (short)0, acc0, false, false);
        acc1 = __builtin_amdgcn_wmma_f32_16x16x32_bf16(false, a.v, false, b1.v, (short)0, acc1, false, false);
        acc2 = __builtin_amdgcn_wmma_f32_16x16x32_bf16(false, a.v, false, b2.v, (short)0, acc2, false, false);
        acc3 = __builtin_amdgcn_wmma_f32_16x16x32_bf16(false, a.v, false, b3.v, (short)0, acc3, false, false);
    }
#pragma unroll
    for (int j = 0; j < 4; j++) {
        v8f acc = (j == 0) ? acc0 : (j == 1) ? acc1 : (j == 2) ? acc2 : acc3;
        int n = tn4 * 64 + j * 16 + l15;
        float bv = bias[n];
#pragma unroll
        for (int v = 0; v < 8; v++) {
            int m = tm * 16 + v + half * 8;
            Y[(size_t)m * F_OUT + n] = acc[v] + bv;
        }
    }
}

// ---------------------------------------------------------------------------
// semantic attention: score += sum_{n,f} q[f]*tanh((X @ kW^T + kb)[n,f])
// same 16x64 WMMA tiling, bf16 inputs, tanh/q epilogue + block reduce
__global__ void __launch_bounds__(256) gemm_tanh_score_kernel(
    const ushort_t* __restrict__ Xb, const ushort_t* __restrict__ KWb,
    const float* __restrict__ kb, const float* __restrict__ q,
    float* __restrict__ score, int M)
{
    __shared__ float sred[256];
    const int K = F_OUT;
    int wave = (int)((blockIdx.x * blockDim.x + threadIdx.x) >> 5);
    int lane = threadIdx.x & 31;
    int tm  = wave >> 2;
    int tn4 = wave & 3;
    float part = 0.f;
    if (tm * 16 < M) {
        int half = lane >> 4;
        int l15  = lane & 15;
        const ushort_t* arow = Xb  + (size_t)(tm * 16 + l15) * K;
        const ushort_t* br0  = KWb + (size_t)(tn4 * 64 + 0  + l15) * K;
        const ushort_t* br1  = KWb + (size_t)(tn4 * 64 + 16 + l15) * K;
        const ushort_t* br2  = KWb + (size_t)(tn4 * 64 + 32 + l15) * K;
        const ushort_t* br3  = KWb + (size_t)(tn4 * 64 + 48 + l15) * K;
        v8f acc0 = {}, acc1 = {}, acc2 = {}, acc3 = {};
        for (int k0 = 0; k0 < K; k0 += 32) {
            Frag16 a, b0, b1, b2, b3;
            a.q[0] = *(const uint4p*)(arow + k0 + half * 16);
            a.q[1] = *(const uint4p*)(arow + k0 + half * 16 + 8);
            int kbo = k0 + half * 16;
            b0.q[0] = *(const uint4p*)(br0 + kbo); b0.q[1] = *(const uint4p*)(br0 + kbo + 8);
            b1.q[0] = *(const uint4p*)(br1 + kbo); b1.q[1] = *(const uint4p*)(br1 + kbo + 8);
            b2.q[0] = *(const uint4p*)(br2 + kbo); b2.q[1] = *(const uint4p*)(br2 + kbo + 8);
            b3.q[0] = *(const uint4p*)(br3 + kbo); b3.q[1] = *(const uint4p*)(br3 + kbo + 8);
            acc0 = __builtin_amdgcn_wmma_f32_16x16x32_bf16(false, a.v, false, b0.v, (short)0, acc0, false, false);
            acc1 = __builtin_amdgcn_wmma_f32_16x16x32_bf16(false, a.v, false, b1.v, (short)0, acc1, false, false);
            acc2 = __builtin_amdgcn_wmma_f32_16x16x32_bf16(false, a.v, false, b2.v, (short)0, acc2, false, false);
            acc3 = __builtin_amdgcn_wmma_f32_16x16x32_bf16(false, a.v, false, b3.v, (short)0, acc3, false, false);
        }
#pragma unroll
        for (int j = 0; j < 4; j++) {
            v8f acc = (j == 0) ? acc0 : (j == 1) ? acc1 : (j == 2) ? acc2 : acc3;
            int n = tn4 * 64 + j * 16 + l15;
            float kbn = kb[n], qn = q[n];
#pragma unroll
            for (int v = 0; v < 8; v++) part += qn * tanhf(acc[v] + kbn);
        }
    }
    sred[threadIdx.x] = part;
    __syncthreads();
    for (int off = 128; off > 0; off >>= 1) {
        if ((int)threadIdx.x < off) sred[threadIdx.x] += sred[threadIdx.x + off];
        __syncthreads();
    }
    if (threadIdx.x == 0) atomicAdd(score, sred[0]);
}

// ---------------------------------------------------------------------------
// a[n,h] = sum_d Hmat[n,h,d] * att[h,d]
__global__ void node_dot_kernel(const float* __restrict__ Hmat,
                                const float* __restrict__ att,
                                float* __restrict__ out, int N)
{
    int t = blockIdx.x * blockDim.x + threadIdx.x;
    if (t >= N * H) return;
    int n = t >> 1, h = t & 1;
    const float4* hp = (const float4*)(Hmat + (size_t)n * F_OUT + h * D);
    const float4* ap = (const float4*)(att + h * D);
    float s = 0.f;
#pragma unroll 8
    for (int i = 0; i < D / 4; i++) {
        float4 a = hp[i], b = ap[i];
        s += a.x * b.x + a.y * b.y + a.z * b.z + a.w * b.w;
    }
    out[t] = s;
}

// alpha[e,h] = leaky_relu(as[src]+ad[dst]); track per-dst max
__global__ void edge_alpha_kernel(const float* __restrict__ as_,
                                  const float* __restrict__ ad_,
                                  const int* __restrict__ ei,
                                  float* __restrict__ alpha,
                                  float* __restrict__ mx)
{
    int e = blockIdx.x * blockDim.x + threadIdx.x;
    if (e >= E_EDGE) return;
    int s = ei[e], d = ei[E_EDGE + e];
#pragma unroll
    for (int h = 0; h < H; h++) {
        float a = as_[s * H + h] + ad_[d * H + h];
        a = a > 0.f ? a : 0.2f * a;
        alpha[e * H + h] = a;
        atomicMaxFloat(&mx[d * H + h], a);
    }
}

// alpha[e,h] <- exp(alpha - mx[dst]); sum[dst,h] += alpha
__global__ void edge_exp_kernel(float* __restrict__ alpha,
                                const int* __restrict__ ei,
                                const float* __restrict__ mx,
                                float* __restrict__ sum)
{
    int e = blockIdx.x * blockDim.x + threadIdx.x;
    if (e >= E_EDGE) return;
    int d = ei[E_EDGE + e];
#pragma unroll
    for (int h = 0; h < H; h++) {
        float w = expf(alpha[e * H + h] - mx[d * H + h]);
        alpha[e * H + h] = w;
        atomicAdd(&sum[d * H + h], w);
    }
}

// out[dst,:] += Hsrc[src,:] * (w / (sum[dst]+eps)); one wave per edge, 8 f/lane
__global__ void __launch_bounds__(256) edge_scatter_kernel(
    const float* __restrict__ Hsrc, const int* __restrict__ ei,
    const float* __restrict__ w, const float* __restrict__ sum,
    float* __restrict__ out)
{
    int wv = (int)((blockIdx.x * blockDim.x + threadIdx.x) >> 5);
    int lane = threadIdx.x & 31;
    if (wv >= E_EDGE) return;
    int s = ei[wv], d = ei[E_EDGE + wv];
    int f0 = lane * 8;
    int h = f0 >> 7;
    float wt = w[wv * H + h] / (sum[d * H + h] + 1e-16f);
    const float4* hp = (const float4*)(Hsrc + (size_t)s * F_OUT + f0);
    float4 v0 = hp[0], v1 = hp[1];
    float* op = out + (size_t)d * F_OUT + f0;
    atomicAdd(op + 0, v0.x * wt); atomicAdd(op + 1, v0.y * wt);
    atomicAdd(op + 2, v0.z * wt); atomicAdd(op + 3, v0.w * wt);
    atomicAdd(op + 4, v1.x * wt); atomicAdd(op + 5, v1.y * wt);
    atomicAdd(op + 6, v1.z * wt); atomicAdd(op + 7, v1.w * wt);
}

// out[i,f] = w0*out_ad[idx[i],f] + w1*out_dd[idx[i],f]; w = softmax(score/N)
__global__ void finalize_kernel(const float* __restrict__ out_ad,
                                const float* __restrict__ out_dd,
                                const float* __restrict__ score,
                                const int* __restrict__ del_idx,
                                float* __restrict__ out)
{
    int i = blockIdx.x;
    int f = threadIdx.x;
    float s0 = score[0] * (1.0f / N_NODE);
    float s1 = score[1] * (1.0f / N_NODE);
    float m = fmaxf(s0, s1);
    float e0 = expf(s0 - m), e1 = expf(s1 - m);
    float inv = 1.0f / (e0 + e1);
    int n = del_idx[i];
    out[(size_t)i * F_OUT + f] = (e0 * inv) * out_ad[(size_t)n * F_OUT + f]
                               + (e1 * inv) * out_dd[(size_t)n * F_OUT + f];
}

// ---------------------------------------------------------------------------
extern "C" void kernel_launch(void* const* d_in, const int* in_sizes, int n_in,
                              void* d_out, int out_size, void* d_ws, size_t ws_size,
                              hipStream_t stream) {
    (void)in_sizes; (void)n_in; (void)out_size; (void)ws_size;
    const float* x_add    = (const float*)d_in[0];
    const float* x_del    = (const float*)d_in[1];
    const float* W_add    = (const float*)d_in[2];
    const float* b_add    = (const float*)d_in[3];
    const float* W_del    = (const float*)d_in[4];
    const float* b_del    = (const float*)d_in[5];
    const float* att_ad_s = (const float*)d_in[6];
    const float* att_ad_d = (const float*)d_in[7];
    // d_in[8..11]: att_da_*, att_aa_* -> dead code (output only uses del branch)
    const float* att_dd_s = (const float*)d_in[12];
    const float* att_dd_d = (const float*)d_in[13];
    const float* k_W      = (const float*)d_in[14];
    const float* k_b      = (const float*)d_in[15];
    const float* q        = (const float*)d_in[16];
    const int*   ei_ad    = (const int*)d_in[17];
    // d_in[18..19]: ei_da, ei_aa -> dead code
    const int*   ei_dd    = (const int*)d_in[20];
    const int*   del_idx  = (const int*)d_in[21];

    float* ws = (float*)d_ws;
    size_t o = 0;
    float* ha       = ws + o; o += (size_t)N_NODE * F_OUT;
    float* hd       = ws + o; o += (size_t)N_NODE * F_OUT;
    float* out_ad   = ws + o; o += (size_t)N_NODE * F_OUT;   // zero region start
    float* out_dd   = ws + o; o += (size_t)N_NODE * F_OUT;
    float* sum_ad   = ws + o; o += (size_t)N_NODE * H;
    float* sum_dd   = ws + o; o += (size_t)N_NODE * H;
    float* score    = ws + o; o += 16;                       // zero region end
    float* mx_ad    = ws + o; o += (size_t)N_NODE * H;       // -inf region
    float* mx_dd    = ws + o; o += (size_t)N_NODE * H;
    float* as_ad    = ws + o; o += (size_t)N_NODE * H;
    float* ad_ad    = ws + o; o += (size_t)N_NODE * H;
    float* as_dd    = ws + o; o += (size_t)N_NODE * H;
    float* ad_dd    = ws + o; o += (size_t)N_NODE * H;
    float* alpha_ad = ws + o; o += (size_t)E_EDGE * H;
    float* alpha_dd = ws + o; o += (size_t)E_EDGE * H;
    // bf16 buffers (allocated in float units; 2 halves per float)
    ushort_t* xb_add  = (ushort_t*)(ws + o); o += (size_t)N_NODE * K_IN / 2;
    ushort_t* xb_del  = (ushort_t*)(ws + o); o += (size_t)N_NODE * K_IN / 2;
    ushort_t* wb_add  = (ushort_t*)(ws + o); o += (size_t)F_OUT * K_IN / 2;
    ushort_t* wb_del  = (ushort_t*)(ws + o); o += (size_t)F_OUT * K_IN / 2;
    ushort_t* kwb     = (ushort_t*)(ws + o); o += (size_t)F_OUT * F_OUT / 2;
    ushort_t* outbf_ad = (ushort_t*)(ws + o); o += (size_t)N_NODE * F_OUT / 2;
    ushort_t* outbf_dd = (ushort_t*)(ws + o); o += (size_t)N_NODE * F_OUT / 2;

    const size_t zeroCount = 2 * (size_t)N_NODE * F_OUT + 2 * (size_t)N_NODE * H + 16;
    const size_t infCount  = 2 * (size_t)N_NODE * H;
    fill_u32_kernel<<<2048, 256, 0, stream>>>((unsigned int*)out_ad, 0u, zeroCount);
    fill_u32_kernel<<<256, 256, 0, stream>>>((unsigned int*)mx_ad, 0xFF800000u, infCount);

    // one-time f32 -> bf16 conversions
    cvt_bf16_kernel<<<2048, 256, 0, stream>>>(x_add, xb_add, (size_t)N_NODE * K_IN);
    cvt_bf16_kernel<<<2048, 256, 0, stream>>>(x_del, xb_del, (size_t)N_NODE * K_IN);
    cvt_bf16_kernel<<<96, 256, 0, stream>>>(W_add, wb_add, (size_t)F_OUT * K_IN);
    cvt_bf16_kernel<<<96, 256, 0, stream>>>(W_del, wb_del, (size_t)F_OUT * K_IN);
    cvt_bf16_kernel<<<32, 256, 0, stream>>>(k_W, kwb, (size_t)F_OUT * F_OUT);

    // projections: (20000/16)*(256/64) = 5000 waves -> 625 blocks
    gemm_bf_kernel<<<625, 256, 0, stream>>>(xb_add, wb_add, b_add, ha, N_NODE, K_IN);
    gemm_bf_kernel<<<625, 256, 0, stream>>>(xb_del, wb_del, b_del, hd, N_NODE, K_IN);

    // per-node attention dots
    int ndBlocks = (N_NODE * H + 255) / 256;
    node_dot_kernel<<<ndBlocks, 256, 0, stream>>>(ha, att_ad_s, as_ad, N_NODE);
    node_dot_kernel<<<ndBlocks, 256, 0, stream>>>(hd, att_ad_d, ad_ad, N_NODE);
    node_dot_kernel<<<ndBlocks, 256, 0, stream>>>(hd, att_dd_s, as_dd, N_NODE);
    node_dot_kernel<<<ndBlocks, 256, 0, stream>>>(hd, att_dd_d, ad_dd, N_NODE);

    // segment softmax (max -> exp/sum -> scatter)
    int eBlocks = (E_EDGE + 255) / 256;
    edge_alpha_kernel<<<eBlocks, 256, 0, stream>>>(as_ad, ad_ad, ei_ad, alpha_ad, mx_ad);
    edge_alpha_kernel<<<eBlocks, 256, 0, stream>>>(as_dd, ad_dd, ei_dd, alpha_dd, mx_dd);
    edge_exp_kernel<<<eBlocks, 256, 0, stream>>>(alpha_ad, ei_ad, mx_ad, sum_ad);
    edge_exp_kernel<<<eBlocks, 256, 0, stream>>>(alpha_dd, ei_dd, mx_dd, sum_dd);
    edge_scatter_kernel<<<E_EDGE / 8, 256, 0, stream>>>(ha, ei_ad, alpha_ad, sum_ad, out_ad);
    edge_scatter_kernel<<<E_EDGE / 8, 256, 0, stream>>>(hd, ei_dd, alpha_dd, sum_dd, out_dd);

    // relu (in place) + bf16 copies (out_ad/out_dd and outbf_ad/outbf_dd contiguous)
    relu_cvt_kernel<<<2048, 256, 0, stream>>>(out_ad, outbf_ad, 2 * (size_t)N_NODE * F_OUT);

    // semantic attention scores (WMMA bf16 + tanh epilogue)
    gemm_tanh_score_kernel<<<625, 256, 0, stream>>>(outbf_ad, kwb, k_b, q, score + 0, N_NODE);
    gemm_tanh_score_kernel<<<625, 256, 0, stream>>>(outbf_dd, kwb, k_b, q, score + 1, N_NODE);

    // softmax over 2 metapaths + gather del_idx
    finalize_kernel<<<N_SEL, 256, 0, stream>>>(out_ad, out_dd, score, del_idx, (float*)d_out);
}
